// CovaBlock_8589934592087
// MI455X (gfx1250) — matrix operand order repaired and checked by reading, not compiled, not run on hardware
//
#include <hip/hip_runtime.h>
#include <hip/hip_bf16.h>
#include <stdint.h>

// ---- problem constants (from reference) ----
#define CC    256      // channels
#define KK    16       // classes
#define SHOT  5
#define HW    441      // 21*21
#define NSUP  2205     // SHOT*HW
#define NPAD  2208     // padded to multiple of 32 (69 * 32)
#define NKS_COV 69     // NPAD/32
#define BB    64       // batch
#define LL    441
#define BL    28224    // BB*LL
#define CTILES 1764    // BL/16
#define CGROUPS 441    // BL/64  (4 column tiles per wave)
#define INV_NM1 (1.0f/2204.0f)

typedef __attribute__((ext_vector_type(16))) __bf16 v16bf;
typedef __attribute__((ext_vector_type(8)))  float  v8f;

struct Frag32 { uint4 lo, hi; };  // 32 bytes -> one v16bf operand

static __device__ __forceinline__ v16bf make_frag(uint4 lo, uint4 hi) {
  Frag32 f; f.lo = lo; f.hi = hi;
  return __builtin_bit_cast(v16bf, f);
}

// float -> bf16 bits, round-to-nearest-even
static __device__ __forceinline__ unsigned short f2bf(float x) {
  unsigned u = __float_as_uint(x);
  u += 0x7FFFu + ((u >> 16) & 1u);
  return (unsigned short)(u >> 16);
}

// unpack 8 packed bf16 (uint4) -> 8 f32
static __device__ __forceinline__ void bf8_to_f32(uint4 u, float* f) {
  f[0] = __uint_as_float(u.x << 16); f[1] = __uint_as_float(u.x & 0xFFFF0000u);
  f[2] = __uint_as_float(u.y << 16); f[3] = __uint_as_float(u.y & 0xFFFF0000u);
  f[4] = __uint_as_float(u.z << 16); f[5] = __uint_as_float(u.z & 0xFFFF0000u);
  f[6] = __uint_as_float(u.w << 16); f[7] = __uint_as_float(u.w & 0xFFFF0000u);
}

// ---------------------------------------------------------------------------
// Kernel 1: per-(k,c) mean over N; write centered bf16 Sc[k][c][n], zero-pad.
// grid = KK*CC blocks of 256 threads.
__global__ void k_prep_support(const float* __restrict__ sup,
                               unsigned short* __restrict__ Sc) {
  const int kc = blockIdx.x;
  const int k  = kc >> 8;           // / CC
  const int c  = kc & 255;
  const int t  = threadIdx.x;
  __shared__ float sm[256];

  float sum = 0.0f;
  for (int n = t; n < NSUP; n += 256) {
    int s = n / HW, p = n - s * HW;
    sum += sup[((size_t)(k * SHOT + s) * CC + c) * HW + p];
  }
  sm[t] = sum; __syncthreads();
  for (int s = 128; s > 0; s >>= 1) {
    if (t < s) sm[t] += sm[t + s];
    __syncthreads();
  }
  const float mean = sm[0] * (1.0f / (float)NSUP);

  unsigned short* row = Sc + (size_t)(k * CC + c) * NPAD;
  for (int n = t; n < NSUP; n += 256) {
    int s = n / HW, p = n - s * HW;
    float v = sup[((size_t)(k * SHOT + s) * CC + c) * HW + p] - mean;
    row[n] = f2bf(v);
  }
  if (t < NPAD - NSUP) row[NSUP + t] = 0;  // zero pad
}

// ---------------------------------------------------------------------------
// Kernel 2: per-column L2 norm over channels; write TRANSPOSED qnT[col][c] bf16.
// grid = BL blocks of 256 threads (one thread per channel).
__global__ void k_prep_query(const float* __restrict__ q,
                             unsigned short* __restrict__ qnT) {
  const int col = blockIdx.x;       // col = b*LL + l
  const int b   = col / LL;
  const int l   = col - b * LL;
  const int c   = threadIdx.x;
  __shared__ float sm[256];

  float v = q[((size_t)b * CC + c) * HW + l];
  sm[c] = v * v; __syncthreads();
  for (int s = 128; s > 0; s >>= 1) {
    if (c < s) sm[c] += sm[c + s];
    __syncthreads();
  }
  const float scale = 1.0f / (sqrtf(sm[0]) + 1e-8f);
  qnT[(size_t)col * CC + c] = f2bf(v * scale);
}

// ---------------------------------------------------------------------------
// Kernel 3: cov[k] = Sc * Sc^T / (N-1), bf16 out. One wave per 16x16 tile.
// tasks = KK * 16 * 16 = 4096 waves -> 512 blocks x 256 threads.
__global__ void k_cov(const unsigned short* __restrict__ Sc,
                      unsigned short* __restrict__ cov) {
  const int wid    = (blockIdx.x * blockDim.x + threadIdx.x) >> 5;
  const int k      = wid >> 8;
  const int tile   = wid & 255;
  const int mt     = tile >> 4;
  const int nt     = tile & 15;
  const int lane   = threadIdx.x & 31;
  const int lane15 = lane & 15;
  const int laneHi = lane >> 4;

  const unsigned short* Arow = Sc + (size_t)(k * CC + mt * 16 + lane15) * NPAD;
  const unsigned short* Brow = Sc + (size_t)(k * CC + nt * 16 + lane15) * NPAD;

  v8f acc = {};
  for (int ks = 0; ks < NKS_COV; ++ks) {
    const int kb = ks * 32;
    // A frag (16x32 bf16): VGPR0-3 = K base, VGPR4-7 = K base+16; half-lanes +8
    uint4 alo = *(const uint4*)(Arow + kb + laneHi * 8);
    uint4 ahi = *(const uint4*)(Arow + kb + laneHi * 8 + 16);
    // B frag (32x16 bf16): VGPR0-3 = K base, VGPR4-7 = K base+16; half-lanes +8
    uint4 blo = *(const uint4*)(Brow + kb + laneHi * 8);
    uint4 bhi = *(const uint4*)(Brow + kb + laneHi * 8 + 16);
    v16bf a = make_frag(alo, ahi);
    v16bf b = make_frag(blo, bhi);
    acc = __builtin_amdgcn_wmma_f32_16x16x32_bf16(
        false, a, false, b, (short)0, acc, false, false);
  }

  // C/D layout: lanes0-15 n=lane, m=r; lanes16-31 n=lane-16, m=r+8
  unsigned short* out = cov + (size_t)k * CC * CC + nt * 16 + lane15;
  const int m0 = mt * 16 + laneHi * 8;
#pragma unroll
  for (int r = 0; r < 8; ++r)
    out[(size_t)(m0 + r) * CC] = f2bf(acc[r] * INV_NM1);
}

// ---------------------------------------------------------------------------
// Kernel 4: sim[b,k,l] = qn^T cov_k qn.
// One wave handles one k and FOUR 16-column tiles (64 cols): each cov A-frag
// feeds 4 back-to-back WMMAs (4x reuse, 4x less cov L2 traffic).
// tasks = KK * CGROUPS = 7056 waves -> 882 blocks x 256 threads.
__global__ void k_sim(const unsigned short* __restrict__ cov,
                      const unsigned short* __restrict__ qnT,
                      float* __restrict__ out) {
  const int wid    = (blockIdx.x * blockDim.x + threadIdx.x) >> 5;
  const int k      = wid / CGROUPS;
  const int grp    = wid - k * CGROUPS;
  const int col0   = grp * 64;
  const int lane   = threadIdx.x & 31;
  const int lane15 = lane & 15;
  const int laneHi = lane >> 4;

  const unsigned short* covk = cov + (size_t)k * CC * CC;
  // per-lane column base for tile t is qcol0 + t*16*CC
  const unsigned short* qcol0 = qnT + (size_t)(col0 + lane15) * CC;

  float s0 = 0.0f, s1 = 0.0f, s2 = 0.0f, s3 = 0.0f;

  for (int mt = 0; mt < 16; ++mt) {
    const unsigned short* Arow = covk + (size_t)(mt * 16 + lane15) * CC;
    __builtin_prefetch(Arow + CC * 16, 0, 0);  // next mtile's cov rows

    // qn tiles in C/D layout: column fixed per lane, 8 contiguous channels
    float qf[4][8];
#pragma unroll
    for (int t = 0; t < 4; ++t) {
      uint4 qc = *(const uint4*)(qcol0 + (size_t)t * 16 * CC + mt * 16 + laneHi * 8);
      bf8_to_f32(qc, qf[t]);
    }

    v8f acc0 = {}, acc1 = {}, acc2 = {}, acc3 = {};
#pragma unroll
    for (int ks = 0; ks < 8; ++ks) {
      const int kb = ks * 32;
      uint4 alo = *(const uint4*)(Arow + kb + laneHi * 16);
      uint4 ahi = *(const uint4*)(Arow + kb + laneHi * 16 + 8);
      v16bf a = make_frag(alo, ahi);

      const unsigned short* qk = qcol0 + kb + laneHi * 16;
      v16bf b0 = make_frag(*(const uint4*)(qk),
                           *(const uint4*)(qk + 8));
      v16bf b1 = make_frag(*(const uint4*)(qk + 16 * CC),
                           *(const uint4*)(qk + 16 * CC + 8));
      v16bf b2 = make_frag(*(const uint4*)(qk + 32 * CC),
                           *(const uint4*)(qk + 32 * CC + 8));
      v16bf b3 = make_frag(*(const uint4*)(qk + 48 * CC),
                           *(const uint4*)(qk + 48 * CC + 8));

      acc0 = __builtin_amdgcn_wmma_f32_16x16x32_bf16(
          false, a, false, b0, (short)0, acc0, false, false);
      acc1 = __builtin_amdgcn_wmma_f32_16x16x32_bf16(
          false, a, false, b1, (short)0, acc1, false, false);
      acc2 = __builtin_amdgcn_wmma_f32_16x16x32_bf16(
          false, a, false, b2, (short)0, acc2, false, false);
      acc3 = __builtin_amdgcn_wmma_f32_16x16x32_bf16(
          false, a, false, b3, (short)0, acc3, false, false);
    }

    // diag reduction is linear: s[t] += sum_m qn[m,n] * T[m,n]
    float d0 = 0.0f, d1 = 0.0f, d2 = 0.0f, d3 = 0.0f;
#pragma unroll
    for (int r = 0; r < 8; ++r) {
      d0 = fmaf(qf[0][r], acc0[r], d0);
      d1 = fmaf(qf[1][r], acc1[r], d1);
      d2 = fmaf(qf[2][r], acc2[r], d2);
      d3 = fmaf(qf[3][r], acc3[r], d3);
    }
    s0 += d0; s1 += d1; s2 += d2; s3 += d3;
  }

  // combine row-halves (lanes j and j+16 hold complementary m-ranges)
  s0 += __shfl_xor(s0, 16, 32);
  s1 += __shfl_xor(s1, 16, 32);
  s2 += __shfl_xor(s2, 16, 32);
  s3 += __shfl_xor(s3, 16, 32);

  if (lane < 16) {
    float sv[4] = {s0, s1, s2, s3};
#pragma unroll
    for (int t = 0; t < 4; ++t) {
      const int g = col0 + t * 16 + lane;
      const int b = g / LL;
      const int l = g - b * LL;
      out[(size_t)b * (KK * LL) + k * LL + l] = sv[t];
    }
  }
}

// ---------------------------------------------------------------------------
extern "C" void kernel_launch(void* const* d_in, const int* in_sizes, int n_in,
                              void* d_out, int out_size, void* d_ws, size_t ws_size,
                              hipStream_t stream) {
  (void)in_sizes; (void)n_in; (void)out_size; (void)ws_size;
  const float* query   = (const float*)d_in[0];  // (64,256,21,21)
  const float* support = (const float*)d_in[1];  // (16,5,256,21,21)
  float* out = (float*)d_out;                    // (64, 16*441)

  // workspace layout (bf16 bit patterns as ushort), all chunks 256B-aligned
  unsigned short* Sc  = (unsigned short*)d_ws;                 // KK*CC*NPAD
  unsigned short* qnT = Sc  + (size_t)KK * CC * NPAD;          // BL*CC
  unsigned short* cov = qnT + (size_t)BL * CC;                 // KK*CC*CC

  k_prep_support<<<KK * CC, 256, 0, stream>>>(support, Sc);
  k_prep_query<<<BL, 256, 0, stream>>>(query, qnT);
  k_cov<<<(KK * 256) / 8, 256, 0, stream>>>(Sc, cov);           // 512 blocks
  k_sim<<<(KK * CGROUPS) / 8, 256, 0, stream>>>(cov, qnT, out); // 882 blocks
}
